// TI_GATWithEdgeAttrs_23244363006444
// MI455X (gfx1250) — compile-verified, matrix operand order checked by reading, not run on hardware
//
#include <hip/hip_runtime.h>

#define F_IN        128
#define EDGE_DIM    16
#define H_HEADS     4
#define DHEAD       8
#define HID         32
#define NUM_GRAPHS  64
#define NUM_CLASSES 4
#define NEG_SLOPE   0.2f

typedef __attribute__((ext_vector_type(16))) _Float16 v16h;
typedef __attribute__((ext_vector_type(8)))  _Float16 v8h;
typedef __attribute__((ext_vector_type(8)))  float    v8f;

// Order-preserving float <-> uint encoding (for atomicMax on signed floats).
__device__ __forceinline__ unsigned fenc(float f) {
  unsigned u = __float_as_uint(f);
  return (u & 0x80000000u) ? ~u : (u | 0x80000000u);
}
__device__ __forceinline__ float fdec(unsigned u) {
  return (u & 0x80000000u) ? __uint_as_float(u & 0x7fffffffu)
                           : __uint_as_float(~u);
}

__device__ __forceinline__ float4 biasrelu4(float4 v, const float* __restrict__ b) {
  float4 r;
  r.x = fmaxf(v.x + b[0], 0.f);
  r.y = fmaxf(v.y + b[1], 0.f);
  r.z = fmaxf(v.z + b[2], 0.f);
  r.w = fmaxf(v.w + b[3], 0.f);
  return r;
}

// ------------------------------------------------------------------ zero ----
__global__ void k_zero(unsigned* __restrict__ p, long n) {
  long i = (long)blockIdx.x * blockDim.x + threadIdx.x;
  if (i < n) p[i] = 0u;
}

// ---------------------------------------------------------- WMMA GEMM -------
// Hout[nrows x 32] = act(A[nrows x K]) @ W[K x 32]; act = relu(a + abias[k])
// when BIAS (compile-time). W is staged once per block into LDS as f16,
// transposed (WT[col][k], pitch K+8 -> 16B-aligned rows), so B fragments are
// ds_load_b128s and A fragments are 4x global_load_b128 per lane per K-step.
// A-frag layout (ISA 7.12.2, 16-bit A 16x32): lane m=lane&15 is row M=m;
// lanes 0-15 hold K in {0..7,16..23}, lanes 16-31 hold K in {8..15,24..31};
// B (32x16) mirrors with N on lanes. Assumes nrows % 16 == 0 (50000 = 16*3125).
template <int K, bool BIAS>
__global__ void k_gemm_wmma(const float* __restrict__ A, const float* __restrict__ W,
                            float* __restrict__ Hout, int nrows,
                            const float* __restrict__ abias) {
  constexpr int PITCH = K + 8;               // halfwords; PITCH*2 % 16 == 0
  __shared__ _Float16 WT[HID * PITCH];
#pragma unroll
  for (int idx = threadIdx.x; idx < HID * K; idx += 256) {
    const int k = idx >> 5, col = idx & 31;  // coalesced read of W[k][col]
    WT[col * PITCH + k] = (_Float16)W[idx];
  }
  __syncthreads();

  const int lane = threadIdx.x & 31;
  const int wave = threadIdx.x >> 5;
  const int row0 = (blockIdx.x * 8 + wave) * 16;
  if (row0 >= nrows) return;                 // wave-uniform: EXEC stays all-1s
  const int m   = lane & 15;
  const int kh8 = (lane >> 4) << 3;          // 0 for lanes 0-15, 8 for 16-31

  const float*    arow = A + (long)(row0 + m) * K;
  const _Float16* wt0  = &WT[m * PITCH];
  const _Float16* wt1  = &WT[(16 + m) * PITCH];

  v8f c0 = {}; v8f c1 = {};
#pragma unroll
  for (int k0 = 0; k0 < K; k0 += 32) {
    const int kb = k0 + kh8;
    float4 a0 = *(const float4*)(arow + kb);
    float4 a1 = *(const float4*)(arow + kb + 4);
    float4 a2 = *(const float4*)(arow + kb + 16);
    float4 a3 = *(const float4*)(arow + kb + 20);
    if constexpr (BIAS) {
      a0 = biasrelu4(a0, abias + kb);
      a1 = biasrelu4(a1, abias + kb + 4);
      a2 = biasrelu4(a2, abias + kb + 16);
      a3 = biasrelu4(a3, abias + kb + 20);
    }
    v16h a;
    a[0]  = (_Float16)a0.x; a[1]  = (_Float16)a0.y; a[2]  = (_Float16)a0.z; a[3]  = (_Float16)a0.w;
    a[4]  = (_Float16)a1.x; a[5]  = (_Float16)a1.y; a[6]  = (_Float16)a1.z; a[7]  = (_Float16)a1.w;
    a[8]  = (_Float16)a2.x; a[9]  = (_Float16)a2.y; a[10] = (_Float16)a2.z; a[11] = (_Float16)a2.w;
    a[12] = (_Float16)a3.x; a[13] = (_Float16)a3.y; a[14] = (_Float16)a3.z; a[15] = (_Float16)a3.w;

    const v8h bl0 = *(const v8h*)(wt0 + kb);        // ds_load_b128 (16B aligned)
    const v8h bh0 = *(const v8h*)(wt0 + kb + 16);
    const v8h bl1 = *(const v8h*)(wt1 + kb);
    const v8h bh1 = *(const v8h*)(wt1 + kb + 16);
    v16h b0, b1;
#pragma unroll
    for (int i = 0; i < 8; ++i) {
      b0[i] = bl0[i]; b0[8 + i] = bh0[i];
      b1[i] = bl1[i]; b1[8 + i] = bh1[i];
    }
    c0 = __builtin_amdgcn_wmma_f32_16x16x32_f16(false, a, false, b0, (short)0, c0, false, false);
    c1 = __builtin_amdgcn_wmma_f32_16x16x32_f16(false, a, false, b1, (short)0, c1, false, false);
  }
#pragma unroll
  for (int r = 0; r < 8; ++r) {              // C/D: VGPR r -> M=r (+8 hi half)
    const long row = row0 + r + kh8;
    Hout[row * HID + m]      = c0[r];
    Hout[row * HID + 16 + m] = c1[r];
  }
}

// -------------------------------------------------- per-node attention ------
__global__ void k_node_attn(const float* __restrict__ Hn,
                            const float* __restrict__ att_src,
                            const float* __restrict__ att_dst,
                            float* __restrict__ a_s, float* __restrict__ a_d, int N) {
  int n = blockIdx.x * blockDim.x + threadIdx.x;
  if (n >= N) return;
  const float* hr = Hn + (long)n * HID;
#pragma unroll
  for (int h = 0; h < H_HEADS; ++h) {
    float s1 = 0.f, s2 = 0.f;
#pragma unroll
    for (int d = 0; d < DHEAD; ++d) {
      float v = hr[h * DHEAD + d];
      s1 += v * att_src[h * DHEAD + d];
      s2 += v * att_dst[h * DHEAD + d];
    }
    a_s[n * 4 + h] = s1;
    a_d[n * 4 + h] = s2;
  }
}

// -------------------- edge pass A: raw alpha (with folded a_e) + seg-max ----
// a_e[e,h] = edge_attr[e,:] @ M[:,h],  M[k,h] = sum_d We[k,h*8+d]*att_edge[h,d]
__global__ void k_edge_alpha_max(const int* __restrict__ src, const int* __restrict__ dst,
                                 const float* __restrict__ eattr,
                                 const float* __restrict__ We, const float* __restrict__ att_edge,
                                 const float* __restrict__ a_s, const float* __restrict__ a_d,
                                 float* __restrict__ alpha, unsigned* __restrict__ amax, long E) {
  __shared__ float Msh[EDGE_DIM][H_HEADS];
  if (threadIdx.x < EDGE_DIM * H_HEADS) {
    int k = threadIdx.x >> 2, h = threadIdx.x & 3;
    float acc = 0.f;
#pragma unroll
    for (int d = 0; d < DHEAD; ++d) acc += We[k * HID + h * DHEAD + d] * att_edge[h * DHEAD + d];
    Msh[k][h] = acc;
  }
  __syncthreads();
  long e = (long)blockIdx.x * blockDim.x + threadIdx.x;
  if (e >= E) return;
  int s = src[e], d = dst[e];
  const float4* eap = (const float4*)(eattr + e * EDGE_DIM);  // 64B row, b128s
  float ea[EDGE_DIM];
  float4 q;
  q = eap[0]; ea[0]  = q.x; ea[1]  = q.y; ea[2]  = q.z; ea[3]  = q.w;
  q = eap[1]; ea[4]  = q.x; ea[5]  = q.y; ea[6]  = q.z; ea[7]  = q.w;
  q = eap[2]; ea[8]  = q.x; ea[9]  = q.y; ea[10] = q.z; ea[11] = q.w;
  q = eap[3]; ea[12] = q.x; ea[13] = q.y; ea[14] = q.z; ea[15] = q.w;
#pragma unroll
  for (int h = 0; h < H_HEADS; ++h) {
    float ae = 0.f;
#pragma unroll
    for (int k = 0; k < EDGE_DIM; ++k) ae += ea[k] * Msh[k][h];
    float al = a_s[(long)s * 4 + h] + a_d[(long)d * 4 + h] + ae;
    al = al > 0.f ? al : NEG_SLOPE * al;          // leaky relu
    alpha[e * 4 + h] = al;
    atomicMax(&amax[(long)d * 4 + h], fenc(al));
  }
}

// -------------------- edge pass B: exp(alpha - max) + seg-sum ---------------
__global__ void k_edge_exp_sum(const int* __restrict__ dst, float* __restrict__ alpha,
                               const unsigned* __restrict__ amax,
                               float* __restrict__ denom, long E) {
  long e = (long)blockIdx.x * blockDim.x + threadIdx.x;
  if (e >= E) return;
  int d = dst[e];
  float4 al = *(const float4*)(alpha + e * 4);          // 16B aligned
  uint4  mu = *(const uint4*)(amax + (long)d * 4);      // 16B aligned
  float4 ex;
  ex.x = expf(al.x - fdec(mu.x));
  ex.y = expf(al.y - fdec(mu.y));
  ex.z = expf(al.z - fdec(mu.z));
  ex.w = expf(al.w - fdec(mu.w));
  *(float4*)(alpha + e * 4) = ex;                       // reuse buffer for ex
  atomicAdd(&denom[(long)d * 4 + 0], ex.x);
  atomicAdd(&denom[(long)d * 4 + 1], ex.y);
  atomicAdd(&denom[(long)d * 4 + 2], ex.z);
  atomicAdd(&denom[(long)d * 4 + 3], ex.w);
}

// -------------------- edge pass C: wave-per-edge weighted scatter-add -------
__global__ void k_edge_aggregate(const int* __restrict__ src, const int* __restrict__ dst,
                                 const float* __restrict__ alpha, const float* __restrict__ denom,
                                 const float* __restrict__ Hn, float* __restrict__ agg, long E) {
  long t = (long)blockIdx.x * blockDim.x + threadIdx.x;
  long e = t >> 5;
  int lane = (int)(t & 31);
  if (e >= E) return;
  int s = src[e], d = dst[e];
  int h = lane >> 3;                               // head = lane/8
  float a = alpha[e * 4 + h] / (denom[(long)d * 4 + h] + 1e-16f);
  float v = Hn[(long)s * HID + lane];              // coalesced 128B row read
  atomicAdd(&agg[(long)d * HID + lane], a * v);    // coalesced 32-lane atomics
}

// -------------------- pool: relu(agg+b) -> per-graph segment max ------------
__global__ void k_pool_max(const float* __restrict__ agg, const float* __restrict__ bias,
                           const int* __restrict__ batch, unsigned* __restrict__ pool, int N) {
  long t = (long)blockIdx.x * blockDim.x + threadIdx.x;
  if (t >= (long)N * HID) return;
  int n = (int)(t >> 5), c = (int)(t & 31);
  float v = agg[t] + bias[c];
  v = v > 0.f ? v : 0.f;                           // >=0: raw bits are monotone
  atomicMax(&pool[(long)batch[n] * HID + c], __float_as_uint(v));
}

// -------------------- tiny MLP head: [64,32]->relu[64,16]->[64,4] -----------
__global__ void k_mlp(const unsigned* __restrict__ pool,
                      const float* __restrict__ fc1w, const float* __restrict__ fc1b,
                      const float* __restrict__ fc2w, const float* __restrict__ fc2b,
                      float* __restrict__ out) {
  int g = threadIdx.x;
  if (g >= NUM_GRAPHS) return;
  float x[HID];
#pragma unroll
  for (int k = 0; k < HID; ++k) x[k] = __uint_as_float(pool[g * HID + k]);
  float z[16];
  for (int j = 0; j < 16; ++j) {
    float acc = fc1b[j];
    for (int k = 0; k < HID; ++k) acc += x[k] * fc1w[k * 16 + j];
    z[j] = acc > 0.f ? acc : 0.f;
  }
  for (int c = 0; c < NUM_CLASSES; ++c) {
    float acc = fc2b[c];
    for (int j = 0; j < 16; ++j) acc += z[j] * fc2w[j * NUM_CLASSES + c];
    out[g * NUM_CLASSES + c] = acc;
  }
}

// ---------------------------------------------------------------------------
extern "C" void kernel_launch(void* const* d_in, const int* in_sizes, int n_in,
                              void* d_out, int out_size, void* d_ws, size_t ws_size,
                              hipStream_t stream) {
  const float* x    = (const float*)d_in[0];
  const int*   ei   = (const int*)d_in[1];
  const float* ea   = (const float*)d_in[2];
  const int*   batch= (const int*)d_in[3];
  const float* W1   = (const float*)d_in[4];
  const float* as1  = (const float*)d_in[5];
  const float* ad1  = (const float*)d_in[6];
  const float* We1  = (const float*)d_in[7];
  const float* ae1  = (const float*)d_in[8];
  const float* b1   = (const float*)d_in[9];
  const float* W2   = (const float*)d_in[10];
  const float* as2  = (const float*)d_in[11];
  const float* ad2  = (const float*)d_in[12];
  const float* We2  = (const float*)d_in[13];
  const float* ae2  = (const float*)d_in[14];
  const float* b2   = (const float*)d_in[15];
  const float* fc1w = (const float*)d_in[16];
  const float* fc1b = (const float*)d_in[17];
  const float* fc2w = (const float*)d_in[18];
  const float* fc2b = (const float*)d_in[19];
  float* out = (float*)d_out;

  const int  N = in_sizes[0] / F_IN;      // 50000
  const long E = in_sizes[1] / 2;         // 1.6M
  const int* src = ei;
  const int* dst = ei + E;

  // workspace layout (agg|amax|denom contiguous so one zero call covers them)
  float*    Hbuf  = (float*)d_ws;                         // N*32 f32
  float*    agg   = Hbuf + (long)N * HID;                 // N*32 f32
  unsigned* amax  = (unsigned*)(agg + (long)N * HID);     // N*4  u32
  float*    denom = (float*)(amax + (long)N * 4);         // N*4  f32
  float*    a_s   = denom + (long)N * 4;                  // N*4
  float*    a_d   = a_s + (long)N * 4;                    // N*4
  float*    alpha = a_d + (long)N * 4;                    // E*4
  unsigned* pool  = (unsigned*)(alpha + E * 4);           // 64*32

  const long zlen = (long)N * (HID + 8);                  // agg+amax+denom
  const int  B = 256;
  auto cdiv = [](long a, long b) { return (unsigned)((a + b - 1) / b); };

  k_zero<<<cdiv(zlen, B), B, 0, stream>>>((unsigned*)agg, zlen);
  k_zero<<<cdiv(NUM_GRAPHS * HID, B), B, 0, stream>>>(pool, NUM_GRAPHS * HID);

  // ---------------- layer 1 ----------------
  k_gemm_wmma<F_IN, false><<<cdiv(N, 128), B, 0, stream>>>(x, W1, Hbuf, N, nullptr);
  k_node_attn<<<cdiv(N, B), B, 0, stream>>>(Hbuf, as1, ad1, a_s, a_d, N);
  k_edge_alpha_max<<<cdiv(E, B), B, 0, stream>>>(src, dst, ea, We1, ae1, a_s, a_d, alpha, amax, E);
  k_edge_exp_sum<<<cdiv(E, B), B, 0, stream>>>(dst, alpha, amax, denom, E);
  k_edge_aggregate<<<cdiv(E * 32, B), B, 0, stream>>>(src, dst, alpha, denom, Hbuf, agg, E);

  // ---------------- layer 2 (relu(agg+b1) fused into GEMM A-load) ----------
  k_gemm_wmma<HID, true><<<cdiv(N, 128), B, 0, stream>>>(agg, W2, Hbuf, N, b1);
  k_zero<<<cdiv(zlen, B), B, 0, stream>>>((unsigned*)agg, zlen);
  k_node_attn<<<cdiv(N, B), B, 0, stream>>>(Hbuf, as2, ad2, a_s, a_d, N);
  k_edge_alpha_max<<<cdiv(E, B), B, 0, stream>>>(src, dst, ea, We2, ae2, a_s, a_d, alpha, amax, E);
  k_edge_exp_sum<<<cdiv(E, B), B, 0, stream>>>(dst, alpha, amax, denom, E);
  k_edge_aggregate<<<cdiv(E * 32, B), B, 0, stream>>>(src, dst, alpha, denom, Hbuf, agg, E);

  // ---------------- pool + MLP head ----------------
  k_pool_max<<<cdiv((long)N * HID, B), B, 0, stream>>>(agg, b2, batch, pool, N);
  k_mlp<<<1, 64, 0, stream>>>(pool, fc1w, fc1b, fc2w, fc2b, out);
}
